// MultiHeadLatentAttention_31722628448847
// MI455X (gfx1250) — compile-verified, matrix operand order checked by reading, not run on hardware
//
#include <hip/hip_runtime.h>

// ---------------------------------------------------------------------------
// MLA forward for MI455X (gfx1250), wave32 + v_wmma_f32_16x16x32_bf16.
// Pipeline: fp32->bf16 convert (Wq pre-scaled by 1/sqrt(128)); bf16 GEMMs
// (q, lk, lv^T) with async global->LDS double-buffered staging;
// flash-attention per (head, latent-group) with DPP-based online softmax;
// final fp32 GEMM with Wo.
// ---------------------------------------------------------------------------

typedef __attribute__((ext_vector_type(16))) __bf16        v16bf;
typedef __attribute__((ext_vector_type(8)))  float         v8f;
typedef __attribute__((ext_vector_type(2)))  unsigned int  u32x2;
typedef __attribute__((ext_vector_type(4)))  unsigned int  u32x4;
typedef __attribute__((ext_vector_type(8)))  unsigned int  u32x8;

__device__ __forceinline__ unsigned short f32_to_bf16(float f) {
    unsigned int u = __float_as_uint(f);
    u += 0x7FFFu + ((u >> 16) & 1u);          // round-to-nearest-even
    return (unsigned short)(u >> 16);
}

// Build a 16xbf16 WMMA fragment from two 16-byte chunks.
__device__ __forceinline__ v16bf make_frag(u32x4 lo, u32x4 hi) {
    u32x8 c;
    c[0] = lo[0]; c[1] = lo[1]; c[2] = lo[2]; c[3] = lo[3];
    c[4] = hi[0]; c[5] = hi[1]; c[6] = hi[2]; c[7] = hi[3];
    return __builtin_bit_cast(v16bf, c);
}

// Per-lane fragment load: row base pointer, hk = lane>>4 selects K halves.
// K elements covered: [hk*8 .. hk*8+7] and [16+hk*8 .. 16+hk*8+7].
__device__ __forceinline__ v16bf frag_row(const unsigned short* row, int hk) {
    u32x4 lo = *(const u32x4*)(row + hk * 8);
    u32x4 hi = *(const u32x4*)(row + 16 + hk * 8);
    return make_frag(lo, hi);
}

__device__ __forceinline__ v8f wmma_bf16f32(v16bf a, v16bf b, v8f c) {
    return __builtin_amdgcn_wmma_f32_16x16x32_bf16(
        /*neg_a=*/false, a, /*neg_b=*/false, b,
        /*c_mod=*/(short)0, c, /*reuse_a=*/false, /*reuse_b=*/false);
}

// Async global -> LDS 16-byte copy (per lane). VDST VGPR carries the LDS byte
// offset (low 32 bits of the generic LDS pointer); tracked by ASYNCcnt.
__device__ __forceinline__ void async_ld_b128(unsigned lds_off, const void* gptr) {
    asm volatile("global_load_async_to_lds_b128 %0, %1, off"
                 :: "v"(lds_off), "v"((unsigned long long)(uintptr_t)gptr)
                 : "memory");
}
__device__ __forceinline__ void wait_async0() {
    asm volatile("s_wait_asynccnt 0" ::: "memory");
}
__device__ __forceinline__ unsigned lds_off32(const void* p) {
    return (unsigned)(unsigned long long)(uintptr_t)p;   // low 32 bits give the LDS offset
}

// --- DPP butterfly reduce+broadcast over each 16-lane row ------------------
// xor1 = quad_perm(1,0,3,2)=0xB1; xor2 = quad_perm(2,3,0,1)=0x4E;
// xor4-equiv = ROW_HALF_MIRROR (0x141, valid once quad-uniform);
// xor8-equiv = ROW_MIRROR (0x140, valid once 8-uniform).
template <int CTRL>
__device__ __forceinline__ float dpp_bcast(float x) {
    return __builtin_bit_cast(float,
        __builtin_amdgcn_update_dpp(0, __builtin_bit_cast(int, x),
                                    CTRL, 0xF, 0xF, true));
}
__device__ __forceinline__ float dpp_max16(float x) {
    x = fmaxf(x, dpp_bcast<0xB1>(x));
    x = fmaxf(x, dpp_bcast<0x4E>(x));
    x = fmaxf(x, dpp_bcast<0x141>(x));
    x = fmaxf(x, dpp_bcast<0x140>(x));
    return x;
}
__device__ __forceinline__ float dpp_sum16(float x) {
    x += dpp_bcast<0xB1>(x);
    x += dpp_bcast<0x4E>(x);
    x += dpp_bcast<0x141>(x);
    x += dpp_bcast<0x140>(x);
    return x;
}

__device__ __forceinline__ void store_val(float* p, float v)           { *p = v; }
__device__ __forceinline__ void store_val(unsigned short* p, float v)  { *p = f32_to_bf16(v); }

// ---------------------------------------------------------------------------
// fp32 -> bf16 conversion with optional scale (n multiple of 4)
// ---------------------------------------------------------------------------
__global__ void __launch_bounds__(256)
cvt_f32_bf16(const float* __restrict__ src, unsigned short* __restrict__ dst,
             int n, float scale) {
    int i4 = (blockIdx.x * blockDim.x + threadIdx.x) * 4;
    if (i4 < n) {
        const float* s = src + i4;
        u32x2 w;
        w[0] = ((unsigned)f32_to_bf16(s[1] * scale) << 16) | f32_to_bf16(s[0] * scale);
        w[1] = ((unsigned)f32_to_bf16(s[3] * scale) << 16) | f32_to_bf16(s[2] * scale);
        *(u32x2*)(dst + i4) = w;
    }
}

// ---------------------------------------------------------------------------
// Generic GEMM: C[m][n] = sum_k A[m][k] * W[n][k]   (i.e. C = A * W^T)
// A: MxK bf16 row-major, W: NxK bf16 row-major.
// Output written as C[row*ors + col*ocs] (allows transposed epilogue for lv^T).
// Block tile 128x128, 8 waves of 32x64 each. K slabs (32 wide) staged via
// global_load_async_to_lds_b128 with double buffering: slab k+1 streams into
// LDS while WMMAs consume slab k; one barrier per slab.
// ---------------------------------------------------------------------------
template <typename OutT>
__global__ void __launch_bounds__(256)
gemm_xwT(const unsigned short* __restrict__ A,
         const unsigned short* __restrict__ W,
         OutT* __restrict__ C,
         int M, int N, int K, long ors, long ocs) {
    __shared__ __align__(16) unsigned short As[2][128 * 40]; // 32 K-cols pad 40
    __shared__ __align__(16) unsigned short Ws[2][128 * 40];

    const int tid  = threadIdx.x;
    const int wv   = tid >> 5;
    const int lane = tid & 31;
    const int lm   = lane & 15;
    const int hk   = lane >> 4;
    const int waveM = wv >> 1;      // 0..3 -> 32 rows each
    const int waveN = wv & 1;       // 0..1 -> 64 cols each
    const long mBlock = (long)blockIdx.y * 128;
    const long nBlock = (long)blockIdx.x * 128;

    // Cooperative staging: 256 threads x 16 bf16 = 128x32 slab per array.
    const int lr = tid >> 1;             // 0..127
    const int lc = (tid & 1) * 16;       // 0 or 16
    const unsigned short* gA = A + (mBlock + lr) * (long)K + lc;
    const unsigned short* gW = W + (nBlock + lr) * (long)K + lc;
    unsigned ldsA[2], ldsW[2];
    ldsA[0] = lds_off32(&As[0][lr * 40 + lc]);
    ldsA[1] = lds_off32(&As[1][lr * 40 + lc]);
    ldsW[0] = lds_off32(&Ws[0][lr * 40 + lc]);
    ldsW[1] = lds_off32(&Ws[1][lr * 40 + lc]);

    v8f acc[2][4];
#pragma unroll
    for (int i = 0; i < 2; ++i)
#pragma unroll
        for (int j = 0; j < 4; ++j) acc[i][j] = (v8f)0.0f;

    // Prologue: fill buffer 0 with slab 0.
    async_ld_b128(ldsA[0],      gA);
    async_ld_b128(ldsA[0] + 16, gA + 8);
    async_ld_b128(ldsW[0],      gW);
    async_ld_b128(ldsW[0] + 16, gW + 8);
    wait_async0();
    __syncthreads();

    int cur = 0;
    for (int k0 = 0; k0 < K; k0 += 32) {
        const int nxt = cur ^ 1;
        if (k0 + 32 < K) {               // stream next slab into other buffer
            async_ld_b128(ldsA[nxt],      gA + k0 + 32);
            async_ld_b128(ldsA[nxt] + 16, gA + k0 + 40);
            async_ld_b128(ldsW[nxt],      gW + k0 + 32);
            async_ld_b128(ldsW[nxt] + 16, gW + k0 + 40);
        }

        v16bf af[2], bw[4];
        af[0] = frag_row(&As[cur][(waveM * 32 + lm) * 40], hk);
        af[1] = frag_row(&As[cur][(waveM * 32 + 16 + lm) * 40], hk);
#pragma unroll
        for (int j = 0; j < 4; ++j)
            bw[j] = frag_row(&Ws[cur][(waveN * 64 + j * 16 + lm) * 40], hk);
#pragma unroll
        for (int i = 0; i < 2; ++i)
#pragma unroll
            for (int j = 0; j < 4; ++j)
                acc[i][j] = wmma_bf16f32(af[i], bw[j], acc[i][j]);

        wait_async0();                   // own fills of nxt complete
        __syncthreads();                 // everyone done reading cur + filling nxt
        cur = nxt;
    }

    // C/D layout: VGPR r, lane L -> row r + 8*(L>>4), col L&15.
#pragma unroll
    for (int i = 0; i < 2; ++i)
#pragma unroll
        for (int j = 0; j < 4; ++j)
#pragma unroll
            for (int r = 0; r < 8; ++r) {
                long row = mBlock + waveM * 32 + i * 16 + r + 8 * hk;
                long col = nBlock + waveN * 64 + j * 16 + lm;
                store_val(C + row * ors + col * ocs, acc[i][j][r]);
            }
}

// ---------------------------------------------------------------------------
// Flash attention per (h,g): one wave owns a 16-query tile, 64 keys/round.
// qb  : [T][2048] bf16    (query projections, PRE-SCALED by 1/sqrt(128))
// lkb : [T][512]  bf16    (latent keys, [t][h*32+d])
// lvT : [512][T]  bf16    (latent values transposed, [h*32+d][t])
// omid: [T][2048] bf16    (attention output before Wo)
// ---------------------------------------------------------------------------
__global__ void __launch_bounds__(256)
mla_attn(const unsigned short* __restrict__ qb,
         const unsigned short* __restrict__ lkb,
         const unsigned short* __restrict__ lvT,
         unsigned short* __restrict__ omid) {
    const int T = 2048;
    __shared__ __align__(16) unsigned short Ps[8 * 16 * 72]; // per-wave 16x64 P pad

    const int tid  = threadIdx.x;
    const int wv   = tid >> 5;
    const int lane = tid & 31;
    const int lm   = lane & 15;
    const int hk   = lane >> 4;
    unsigned short* Pw = Ps + wv * (16 * 72);

    const int wid = blockIdx.x * 8 + wv;     // 8192 waves total
    const int qt  = wid & 127;               // query tile (16 rows)
    const int hg  = wid >> 7;
    const int h   = hg >> 2;
    const int g   = hg & 3;
    const int qc  = h * 128 + g * 32;

    // Q fragment: 16 rows x Dl=32 (exactly one WMMA K); scale baked into Wq.
    v16bf qa = frag_row(qb + (long)(qt * 16 + lm) * 2048 + qc, hk);

    float mrow[8], lrow[8];
    v8f o0 = (v8f)0.0f, o1 = (v8f)0.0f;
#pragma unroll
    for (int r = 0; r < 8; ++r) { mrow[r] = -1e30f; lrow[r] = 0.0f; }

    for (int s0 = 0; s0 < T; s0 += 64) {
        // Keys: B[k=d][n=key] = lk[s0+n][d]; lane lm reads key row s0+..+lm.
        const unsigned short* kbase = lkb + (long)(s0 + lm) * 512 + h * 32;
        v16bf bk0 = frag_row(kbase, hk);
        v16bf bk1 = frag_row(kbase + 16 * 512, hk);
        v16bf bk2 = frag_row(kbase + 32 * 512, hk);
        v16bf bk3 = frag_row(kbase + 48 * 512, hk);
        const unsigned short* vbase = lvT + (long)(h * 32 + lm) * T + s0;
        if (s0 + 64 < T) {
            __builtin_prefetch(kbase + 64 * 512, 0, 0);
            __builtin_prefetch(vbase + 64, 0, 0);
        }
        v8f z = (v8f)0.0f;
        v8f sA = wmma_bf16f32(qa, bk0, z);   // keys s0+ 0..15
        v8f sB = wmma_bf16f32(qa, bk1, z);   // keys s0+16..31
        v8f sC = wmma_bf16f32(qa, bk2, z);   // keys s0+32..47
        v8f sD = wmma_bf16f32(qa, bk3, z);   // keys s0+48..63

        // Online softmax in C-layout: VGPR r holds query row r + 8*hk.
#pragma unroll
        for (int r = 0; r < 8; ++r) {
            float x0 = sA[r], x1 = sB[r], x2 = sC[r], x3 = sD[r];
            float t  = fmaxf(fmaxf(x0, x1), fmaxf(x2, x3));
            t = dpp_max16(t);
            float mn = fmaxf(mrow[r], t);
            float p0 = __expf(x0 - mn);
            float p1 = __expf(x1 - mn);
            float p2 = __expf(x2 - mn);
            float p3 = __expf(x3 - mn);
            float rs = (p0 + p1) + (p2 + p3);
            rs = dpp_sum16(rs);
            float corr = __expf(mrow[r] - mn);
            lrow[r] = lrow[r] * corr + rs;
            mrow[r] = mn;
            o0[r] *= corr;
            o1[r] *= corr;
            // Relayout P (C-frag) -> LDS [m][k] for use as A-frag.
            unsigned short* pr = Pw + (r + 8 * hk) * 72;
            pr[lm]      = f32_to_bf16(p0);
            pr[16 + lm] = f32_to_bf16(p1);
            pr[32 + lm] = f32_to_bf16(p2);
            pr[48 + lm] = f32_to_bf16(p3);
        }
        asm volatile("s_wait_dscnt 0" ::: "memory");

        const unsigned short* prow = Pw + lm * 72;
        v16bf pa0 = frag_row(prow, hk);        // P cols  0..31
        v16bf pa1 = frag_row(prow + 32, hk);   // P cols 32..63
        v16bf bv0 = frag_row(vbase, hk);               // keys s0..+31, d=lm
        v16bf bv1 = frag_row(vbase + 16 * T, hk);      // keys s0..+31, d=16+lm
        v16bf bv2 = frag_row(vbase + 32, hk);          // keys s0+32..+63, d=lm
        v16bf bv3 = frag_row(vbase + 16 * T + 32, hk); // keys s0+32..+63, d=16+lm
        o0 = wmma_bf16f32(pa0, bv0, o0);
        o1 = wmma_bf16f32(pa0, bv1, o1);
        o0 = wmma_bf16f32(pa1, bv2, o0);
        o1 = wmma_bf16f32(pa1, bv3, o1);
    }

#pragma unroll
    for (int r = 0; r < 8; ++r) {
        float inv = 1.0f / lrow[r];
        long row = (long)(qt * 16 + r + 8 * hk);
        omid[row * 2048 + qc + lm]      = f32_to_bf16(o0[r] * inv);
        omid[row * 2048 + qc + 16 + lm] = f32_to_bf16(o1[r] * inv);
    }
}

// ---------------------------------------------------------------------------
// Host-side launcher. Inputs: x, Wq, Wk(unused), Wv(unused), Wlk, Wlv, Wo.
// Workspace layout (bf16 staging, 48 MiB total; MiB-aligned offsets):
//   [0,8)   xb      [8,16)  WqB     [16,24) WoB     [24,26) WlkB
//   [26,28) WlvB    [28,36) qbuf    [36,38) lkb     [38,40) lvT    [40,48) omid
// ---------------------------------------------------------------------------
extern "C" void kernel_launch(void* const* d_in, const int* in_sizes, int n_in,
                              void* d_out, int out_size, void* d_ws, size_t ws_size,
                              hipStream_t stream) {
    const float* x   = (const float*)d_in[0];
    const float* Wq  = (const float*)d_in[1];
    const float* Wlk = (const float*)d_in[4];
    const float* Wlv = (const float*)d_in[5];
    const float* Wo  = (const float*)d_in[6];
    float* out = (float*)d_out;

    char* w = (char*)d_ws;
    const size_t MB = 1u << 20;
    unsigned short* xb   = (unsigned short*)(w + 0 * MB);
    unsigned short* WqB  = (unsigned short*)(w + 8 * MB);
    unsigned short* WoB  = (unsigned short*)(w + 16 * MB);
    unsigned short* WlkB = (unsigned short*)(w + 24 * MB);
    unsigned short* WlvB = (unsigned short*)(w + 26 * MB);
    unsigned short* qbuf = (unsigned short*)(w + 28 * MB);
    unsigned short* lkb  = (unsigned short*)(w + 36 * MB);
    unsigned short* lvT  = (unsigned short*)(w + 38 * MB);
    unsigned short* omid = (unsigned short*)(w + 40 * MB);

    const int T = 2048, C = 2048, L = 512;
    const float qscale = 0.0883883476483184f; // 1/sqrt(128), folded into Wq

    cvt_f32_bf16<<<dim3((T * C) / 1024), dim3(256), 0, stream>>>(x,   xb,   T * C, 1.0f);
    cvt_f32_bf16<<<dim3((C * C) / 1024), dim3(256), 0, stream>>>(Wq,  WqB,  C * C, qscale);
    cvt_f32_bf16<<<dim3((C * C) / 1024), dim3(256), 0, stream>>>(Wo,  WoB,  C * C, 1.0f);
    cvt_f32_bf16<<<dim3((L * C) / 1024), dim3(256), 0, stream>>>(Wlk, WlkB, L * C, 1.0f);
    cvt_f32_bf16<<<dim3((L * C) / 1024), dim3(256), 0, stream>>>(Wlv, WlvB, L * C, 1.0f);

    // q = x Wq^T  (bf16, row-major [t][o], pre-scaled)
    gemm_xwT<unsigned short><<<dim3(16, 16), dim3(256), 0, stream>>>(
        xb, WqB, qbuf, T, C, C, (long)C, 1L);
    // lk = x Wlk^T (bf16, [t][h*32+d])
    gemm_xwT<unsigned short><<<dim3(4, 16), dim3(256), 0, stream>>>(
        xb, WlkB, lkb, T, L, C, (long)L, 1L);
    // lv^T = (x Wlv^T)^T (bf16, [h*32+d][t]) via transposed epilogue
    gemm_xwT<unsigned short><<<dim3(4, 16), dim3(256), 0, stream>>>(
        xb, WlvB, lvT, T, L, C, 1L, (long)T);

    // attention: 64 (h,g) pairs x 128 query tiles = 8192 waves / 8 per block
    mla_attn<<<dim3(1024), dim3(256), 0, stream>>>(qbuf, lkb, lvT, omid);

    // out = omid Wo^T (fp32 output)
    gemm_xwT<float><<<dim3(16, 16), dim3(256), 0, stream>>>(
        omid, WoB, out, T, C, C, (long)C, 1L);
}
